// SingleHead_attention_recombinator_25812753449335
// MI455X (gfx1250) — matrix operand order repaired and checked by reading, not compiled, not run on hardware
//
#include <hip/hip_runtime.h>
#include <hip/hip_bf16.h>

typedef __attribute__((ext_vector_type(16))) __bf16 v16bf;
typedef __attribute__((ext_vector_type(8)))  __bf16 v8bf;
typedef __attribute__((ext_vector_type(4)))  __bf16 v4bf;
typedef __attribute__((ext_vector_type(8)))  float  v8f;

#define B_DIM 32768
#define F_DIM 1024
#define H_DIM 512
#define E_DIM 10
#define K_DIM 4
#define C_DIM 10

#define LDS_STRIDE (F_DIM + 8)   // padded bf16 row stride: 2064B, conflict-free b128 reads

// ---- fragment loaders (wave32, v_wmma_f32_16x16x32_bf16 layouts) ----

// A matrix 16x32 (MxK), row-major source with given element stride.
// lanes 0-15: elems = K{0..7, 16..23} of row M=lane; lanes 16-31: K{8..15, 24..31}.
__device__ __forceinline__ v16bf load_frag_a(const __bf16* __restrict__ base, int stride) {
  const int lane = threadIdx.x & 31;
  const __bf16* p = base + (lane & 15) * stride + ((lane >> 4) << 3);
  v8bf lo = *(const v8bf*)(p);
  v8bf hi = *(const v8bf*)(p + 16);
  return __builtin_shufflevector(lo, hi, 0,1,2,3,4,5,6,7,8,9,10,11,12,13,14,15);
}

// B matrix 32x16 (KxN) loaded from row-major B^T (NxK) source.
// lanes 0-15: elems = K{0..7, 16..23} of col N=lane; lanes 16-31: K{8..15, 24..31}.
__device__ __forceinline__ v16bf load_frag_b(const __bf16* __restrict__ base, int stride) {
  const int lane = threadIdx.x & 31;
  const __bf16* p = base + (lane & 15) * stride + ((lane >> 4) << 3);
  v8bf lo = *(const v8bf*)(p);
  v8bf hi = *(const v8bf*)(p + 16);
  return __builtin_shufflevector(lo, hi, 0,1,2,3,4,5,6,7,8,9,10,11,12,13,14,15);
}

// ---- kernel 1: weight conversion/packing ----
__global__ void prep_kernel(const float* __restrict__ Wlin,
                            const float* __restrict__ Wq,
                            const float* __restrict__ Wk,
                            const float* __restrict__ Wv,
                            __bf16* __restrict__ WlinB,
                            __bf16* __restrict__ WqkvB) {
  const int n1 = K_DIM * H_DIM * F_DIM;
  for (int i = blockIdx.x * blockDim.x + threadIdx.x; i < n1;
       i += gridDim.x * blockDim.x) {
    WlinB[i] = (__bf16)Wlin[i];
  }
  // padded [k][mat][n=16][h=512], rows n>=10 are zero
  const int n2 = K_DIM * 3 * 16 * H_DIM;
  for (int j = blockIdx.x * blockDim.x + threadIdx.x; j < n2;
       j += gridDim.x * blockDim.x) {
    const int f  = j & (H_DIM - 1);
    const int n  = (j >> 9) & 15;
    const int km = j >> 13;            // k*3 + mat
    const int m  = km % 3;
    const int k  = km / 3;
    float val = 0.f;
    if (n < E_DIM) {
      const float* W = (m == 0) ? Wq : (m == 1) ? Wk : Wv;
      val = W[(k * E_DIM + n) * H_DIM + f];
    }
    WqkvB[j] = (__bf16)val;
  }
}

// ---- kernel 2: h[b,k,:] = w_k[b,:] @ Wlin[k]^T + blin[k,:]  (bf16 WMMA) ----
__global__ void __launch_bounds__(256, 1)
gemm_h_kernel(const float* __restrict__ w0, const float* __restrict__ w1,
              const float* __restrict__ w2, const float* __restrict__ w3,
              const float* __restrict__ blin,
              const __bf16* __restrict__ WlinB,
              __bf16* __restrict__ hB) {
  extern __shared__ __bf16 smem[];          // 128 x LDS_STRIDE bf16 (~258 KB)
  const int m0   = blockIdx.x * 128;
  const int k    = blockIdx.y;
  const int tid  = threadIdx.x;
  const int lane = tid & 31;
  const int wave = tid >> 5;
  const int wm   = wave & 1;                // 2 waves along M (64 rows each)
  const int wn   = wave >> 1;               // 4 waves along N (32 cols each)

  const float* __restrict__ wk = (k == 0) ? w0 : (k == 1) ? w1 : (k == 2) ? w2 : w3;

  // Stage A tile: 128 x 1024 f32 -> bf16 in LDS (read from HBM exactly once)
  for (int c = tid; c < 128 * 256; c += 256) {
    const int r   = c >> 8;
    const int col = (c & 255) << 2;
    const float4 v = *(const float4*)(wk + (size_t)(m0 + r) * F_DIM + col);
    v4bf pk = { (__bf16)v.x, (__bf16)v.y, (__bf16)v.z, (__bf16)v.w };
    *(v4bf*)(smem + r * LDS_STRIDE + col) = pk;
  }
  __syncthreads();

  const __bf16* __restrict__ Wl = WlinB + (size_t)k * H_DIM * F_DIM;

#pragma unroll 1
  for (int nt = 0; nt < 4; ++nt) {          // sweep N = 512 in 128-wide tiles
    // Opaque barrier: stop LICM from hoisting this sweep's LDS A-fragment
    // reads out of the nt loop (which spills ~4KB/lane to scratch).
    asm volatile("" ::: "memory");

    const int nbase = nt * 128 + wn * 32;
    v8f acc[4][2];
#pragma unroll
    for (int a = 0; a < 4; ++a)
#pragma unroll
      for (int b = 0; b < 2; ++b)
        acc[a][b] = v8f{0.f, 0.f, 0.f, 0.f, 0.f, 0.f, 0.f, 0.f};

#pragma unroll 2
    for (int f0 = 0; f0 < F_DIM; f0 += 32) {
      const v16bf bf0 = load_frag_b(Wl + (size_t)nbase * F_DIM + f0, F_DIM);
      const v16bf bf1 = load_frag_b(Wl + (size_t)(nbase + 16) * F_DIM + f0, F_DIM);
#pragma unroll
      for (int mt = 0; mt < 4; ++mt) {
        const v16bf af = load_frag_a(smem + (wm * 64 + mt * 16) * LDS_STRIDE + f0,
                                     LDS_STRIDE);
        acc[mt][0] = __builtin_amdgcn_wmma_f32_16x16x32_bf16(
            false, af, false, bf0, (short)0, acc[mt][0], false, false);
        acc[mt][1] = __builtin_amdgcn_wmma_f32_16x16x32_bf16(
            false, af, false, bf1, (short)0, acc[mt][1], false, false);
      }
    }

    // epilogue: +blin, store bf16 h  (C/D layout: VGPR i -> M=i(+8), N=lane&15)
#pragma unroll
    for (int mt = 0; mt < 4; ++mt)
#pragma unroll
      for (int ntt = 0; ntt < 2; ++ntt) {
        const int col  = nbase + ntt * 16 + (lane & 15);
        const float bs = blin[k * H_DIM + col];
        const int rb   = m0 + wm * 64 + mt * 16 + ((lane >> 4) << 3);
#pragma unroll
        for (int i = 0; i < 8; ++i)
          hB[(size_t)(rb + i) * (K_DIM * H_DIM) + k * H_DIM + col] =
              (__bf16)(acc[mt][ntt][i] + bs);
      }
  }
}

// ---- kernel 3: q/k/v (WMMA), attention, softmax, logits, outputs ----
__global__ void __launch_bounds__(64, 1)
tail_kernel(const __bf16* __restrict__ hB, const __bf16* __restrict__ WqkvB,
            const float* __restrict__ bq, const float* __restrict__ bk,
            const float* __restrict__ bv, const float* __restrict__ Wfin,
            const float* __restrict__ bfin, float* __restrict__ out) {
  __shared__ float qkv[3][K_DIM][16][16];   // [mat][k][row][col(e, padded)]
  const int b0   = blockIdx.x * 16;
  const int tid  = threadIdx.x;
  const int wave = tid >> 5;
  const int lane = tid & 31;

  // each wave computes q,k,v for two k's over 16 rows: (16x512)x(512x16) WMMA.
  // f0 is the outer accumulation loop so each A fragment is loaded once and
  // shared by the three (q,k,v) WMMAs; 3 accumulators stay live (24 VGPRs).
#pragma unroll 1
  for (int kk = wave * 2; kk < wave * 2 + 2; ++kk) {
    asm volatile("" ::: "memory");
    v8f acc[3];
#pragma unroll
    for (int m = 0; m < 3; ++m)
      acc[m] = v8f{0.f, 0.f, 0.f, 0.f, 0.f, 0.f, 0.f, 0.f};

#pragma unroll 2
    for (int f0 = 0; f0 < H_DIM; f0 += 32) {
      const v16bf af = load_frag_a(
          hB + (size_t)b0 * (K_DIM * H_DIM) + kk * H_DIM + f0, K_DIM * H_DIM);
#pragma unroll
      for (int m = 0; m < 3; ++m) {
        const v16bf bf = load_frag_b(
            WqkvB + (size_t)((kk * 3 + m) * 16) * H_DIM + f0, H_DIM);
        acc[m] = __builtin_amdgcn_wmma_f32_16x16x32_bf16(
            false, af, false, bf, (short)0, acc[m], false, false);
      }
    }
#pragma unroll
    for (int m = 0; m < 3; ++m) {
      const int col = lane & 15;
      const float* __restrict__ bias = (m == 0) ? bq : (m == 1) ? bk : bv;
      const float bs = (col < E_DIM) ? bias[kk * E_DIM + col] : 0.f;
      const int rb = (lane >> 4) << 3;
#pragma unroll
      for (int i = 0; i < 8; ++i)
        qkv[m][kk][rb + i][col] = acc[m][i] + bs;
    }
  }
  __syncthreads();

  if (tid < 16) {
    const int b = b0 + tid;
    float q[K_DIM][E_DIM], ke[K_DIM][E_DIM], v0[E_DIM];
#pragma unroll
    for (int k2 = 0; k2 < K_DIM; ++k2)
#pragma unroll
      for (int e = 0; e < E_DIM; ++e) {
        q[k2][e]  = qkv[0][k2][tid][e];
        ke[k2][e] = qkv[1][k2][tid][e];
      }
#pragma unroll
    for (int e = 0; e < E_DIM; ++e) v0[e] = qkv[2][0][tid][e];

    float attn[E_DIM][E_DIM];
    const float sc = 0.31622776601683794f;  // 1/sqrt(E)
#pragma unroll
    for (int e = 0; e < E_DIM; ++e)
#pragma unroll
      for (int l = 0; l < E_DIM; ++l) {
        float s = 0.f;
#pragma unroll
        for (int k2 = 0; k2 < K_DIM; ++k2) s += q[k2][e] * ke[k2][l];
        attn[e][l] = s * sc;
      }
    // softmax over axis=1 (e), i.e. per column l
#pragma unroll
    for (int l = 0; l < E_DIM; ++l) {
      float mx = attn[0][l];
#pragma unroll
      for (int e = 1; e < E_DIM; ++e) mx = fmaxf(mx, attn[e][l]);
      float s = 0.f;
#pragma unroll
      for (int e = 0; e < E_DIM; ++e) {
        const float p = __expf(attn[e][l] - mx);
        attn[e][l] = p;
        s += p;
      }
      const float inv = 1.f / s;
#pragma unroll
      for (int e = 0; e < E_DIM; ++e) attn[e][l] *= inv;
    }
    // write attn (second output, after outputs block)
    float* __restrict__ oat =
        out + (size_t)B_DIM * C_DIM + (size_t)b * (E_DIM * E_DIM);
#pragma unroll
    for (int e = 0; e < E_DIM; ++e)
#pragma unroll
      for (int l = 0; l < E_DIM; ++l) oat[e * E_DIM + l] = attn[e][l];

    // w0o = att_out[:,0,:]
    float w0o[E_DIM];
#pragma unroll
    for (int n = 0; n < E_DIM; ++n) {
      float s = 0.f;
#pragma unroll
      for (int m2 = 0; m2 < E_DIM; ++m2) s += attn[n][m2] * v0[m2];
      w0o[n] = s;
    }
    // logits -> softmax -> mean over k
    float outc[C_DIM];
#pragma unroll
    for (int c = 0; c < C_DIM; ++c) outc[c] = 0.f;
    for (int k2 = 0; k2 < K_DIM; ++k2) {
      float lg[C_DIM];
      float mx = -3.4e38f;
#pragma unroll
      for (int c = 0; c < C_DIM; ++c) {
        float s = bfin[k2 * C_DIM + c];
#pragma unroll
        for (int e = 0; e < E_DIM; ++e)
          s += w0o[e] * Wfin[(k2 * C_DIM + c) * E_DIM + e];
        lg[c] = s;
        mx = fmaxf(mx, s);
      }
      float ssum = 0.f;
#pragma unroll
      for (int c = 0; c < C_DIM; ++c) {
        lg[c] = __expf(lg[c] - mx);
        ssum += lg[c];
      }
      const float inv = 0.25f / ssum;
#pragma unroll
      for (int c = 0; c < C_DIM; ++c) outc[c] += lg[c] * inv;
    }
#pragma unroll
    for (int c = 0; c < C_DIM; ++c) out[(size_t)b * C_DIM + c] = outc[c];
  }
}

extern "C" void kernel_launch(void* const* d_in, const int* in_sizes, int n_in,
                              void* d_out, int out_size, void* d_ws, size_t ws_size,
                              hipStream_t stream) {
  const float* w0   = (const float*)d_in[0];
  const float* w1   = (const float*)d_in[1];
  const float* w2   = (const float*)d_in[2];
  const float* w3   = (const float*)d_in[3];
  const float* Wlin = (const float*)d_in[4];
  const float* blin = (const float*)d_in[5];
  const float* Wq   = (const float*)d_in[6];
  const float* bq   = (const float*)d_in[7];
  const float* Wk   = (const float*)d_in[8];
  const float* bk   = (const float*)d_in[9];
  const float* Wv   = (const float*)d_in[10];
  const float* bv   = (const float*)d_in[11];
  const float* Wfin = (const float*)d_in[12];
  const float* bfin = (const float*)d_in[13];
  float* out = (float*)d_out;

  char* ws = (char*)d_ws;
  __bf16* WlinB = (__bf16*)ws;                            // 4 MB
  __bf16* WqkvB = (__bf16*)(ws + (4u << 20));             // 192 KB
  __bf16* hB    = (__bf16*)(ws + (5u << 20));             // 128 MB: [b][k*512+h] bf16

  prep_kernel<<<4096, 256, 0, stream>>>(Wlin, Wq, Wk, Wv, WlinB, WqkvB);

  dim3 g2(B_DIM / 128, K_DIM);
  const size_t lds = (size_t)128 * LDS_STRIDE * sizeof(__bf16);  // ~258 KB
  gemm_h_kernel<<<g2, 256, lds, stream>>>(w0, w1, w2, w3, blin, WlinB, hB);

  tail_kernel<<<B_DIM / 16, 64, 0, stream>>>(hB, WqkvB, bq, bk, bv, Wfin, bfin, out);
}
